// TrojectoryPredictor_68109591380154
// MI455X (gfx1250) — compile-verified, matrix operand order checked by reading, not run on hardware
//
#include <hip/hip_runtime.h>

typedef _Float16 v16h __attribute__((ext_vector_type(16)));
typedef float    v8f  __attribute__((ext_vector_type(8)));

#define B_SZ   16
#define N_NODE 128
#define E_SZ   (N_NODE * (N_NODE - 1))     // 16256
#define ROWS   (B_SZ * E_SZ)               // 260096
#define F_OUT  8

__device__ __forceinline__ float sigmoidf_(float x) { return 1.0f / (1.0f + __expf(-x)); }

// y[n0+lane] (lane<16) = sum_k x[k] * W[k*256 + n0+lane], computed via
// v_wmma_f32_16x16x32_f16 with A-row M=0 carrying x, rows 1..15 zero.
// ISA 16-bit A 16x32 layout: lanes 0-15 M=lane hold K {0..7,16..23} (2 halves/VGPR),
// lanes 16-31 M=lane-16 hold K {8..15,24..31}. B 32x16 mirrors this with N=lane&15.
template <int K>
__device__ __forceinline__ float wave_matvec16(const float* xlds,
                                               const float* __restrict__ W,
                                               int n0, int lane) {
    v8f c;
#pragma unroll
    for (int i = 0; i < 8; ++i) c[i] = 0.0f;

    const int sel = (lane >> 4) * 8;       // 0 for lanes 0-15, 8 for lanes 16-31
    const int n   = n0 + (lane & 15);

    for (int kk = 0; kk < K; kk += 32) {
        v16h a, b;
        if ((lane & 15) == 0) {            // M==0 row of A (lanes 0 and 16)
#pragma unroll
            for (int j = 0; j < 8; ++j) {
                a[j]     = (_Float16)xlds[kk + sel + j];
                a[8 + j] = (_Float16)xlds[kk + sel + 16 + j];
            }
        } else {
#pragma unroll
            for (int j = 0; j < 16; ++j) a[j] = (_Float16)0.0f;
        }
#pragma unroll
        for (int j = 0; j < 8; ++j) {
            b[j]     = (_Float16)W[(kk + sel + j) * 256 + n];
            b[8 + j] = (_Float16)W[(kk + sel + 16 + j) * 256 + n];
        }
        c = __builtin_amdgcn_wmma_f32_16x16x32_f16(false, a, false, b,
                                                   (short)0, c, false, false);
    }
    return c[0];   // C VGPR0, lanes 0-15 == row M=0
}

// Single workgroup: mat-vec chain  m2_beta -> logits(8) -> prob(8), written to ws.
__global__ void __launch_bounds__(512)
nri_head_kernel(const float* __restrict__ m2_beta,
                const float* __restrict__ Wi,  const float* __restrict__ bi,
                const float* __restrict__ Wn,  const float* __restrict__ bnb,
                const float* __restrict__ f1W, const float* __restrict__ f1b,
                const float* __restrict__ f2W, const float* __restrict__ f2b,
                const float* __restrict__ f3W, const float* __restrict__ f3b,
                float* __restrict__ head_out)  // [0..7]=logits, [8..15]=prob
{
    __shared__ float xcat[512];
    __shared__ float hid[256];
    __shared__ float o1[256];
    __shared__ float o2[256];
    __shared__ float lgt[8];

    const int tid  = threadIdx.x;
    const int lane = tid & 31;
    const int wave = tid >> 5;     // 0..15
    const int n0   = wave * 16;

    // Both BN-MLP blocks reduce to their beta (identical rows -> h==mean, var==0);
    // x_cat = [m2_beta, m2_beta]
    xcat[tid] = m2_beta[tid & 255];
    __syncthreads();

    float yi = wave_matvec16<512>(xcat, Wi, n0, lane);
    float yn = wave_matvec16<512>(xcat, Wn, n0, lane);
    if (lane < 16) {
        int n = n0 + lane;
        float s = sigmoidf_(yi + bi[n]);
        float t = tanhf(yn + bnb[n]);
        hid[n] = (1.0f - s) * t;           // GRU with zero initial hidden
    }
    __syncthreads();

    float y1 = wave_matvec16<256>(hid, f1W, n0, lane);
    if (lane < 16) { int n = n0 + lane; o1[n] = fmaxf(y1 + f1b[n], 0.0f); }
    __syncthreads();

    float y2 = wave_matvec16<256>(o1, f2W, n0, lane);
    if (lane < 16) { int n = n0 + lane; o2[n] = fmaxf(y2 + f2b[n], 0.0f); }
    __syncthreads();

    if (tid < F_OUT) {
        float acc = f3b[tid];
        for (int k = 0; k < 256; ++k) acc += o2[k] * f3W[k * F_OUT + tid];
        lgt[tid] = acc;
    }
    __syncthreads();

    if (tid == 0) {
        float mx = lgt[0];
#pragma unroll
        for (int j = 1; j < F_OUT; ++j) mx = fmaxf(mx, lgt[j]);
        float e[F_OUT], s = 0.0f;
#pragma unroll
        for (int j = 0; j < F_OUT; ++j) { e[j] = __expf(lgt[j] - mx); s += e[j]; }
        float inv = 1.0f / s;
#pragma unroll
        for (int j = 0; j < F_OUT; ++j) {
            head_out[j]         = lgt[j];      // logits
            head_out[F_OUT + j] = e[j] * inv;  // prob
        }
    }
}

// Bandwidth-bound pass: per row r, g = softmax((logits + gumbel[r]) / tau),
// and broadcast prob. 25 MB total traffic -> ~1.1 us at 23.3 TB/s.
__global__ void __launch_bounds__(256)
nri_gumbel_kernel(const float* __restrict__ gumbel,
                  const float* __restrict__ head,    // uniform: scalar loads
                  float* __restrict__ g_out,
                  float* __restrict__ p_out)
{
    const long r = (long)blockIdx.x * blockDim.x + threadIdx.x;
    if (r >= ROWS) return;

    float lg[F_OUT], pr[F_OUT];
#pragma unroll
    for (int j = 0; j < F_OUT; ++j) { lg[j] = head[j]; pr[j] = head[F_OUT + j]; }

    const float4* g4 = (const float4*)(gumbel + r * F_OUT);
    float4 ga = g4[0], gb = g4[1];

    float t[F_OUT];
    t[0] = (lg[0] + ga.x) * 2.0f;  t[1] = (lg[1] + ga.y) * 2.0f;
    t[2] = (lg[2] + ga.z) * 2.0f;  t[3] = (lg[3] + ga.w) * 2.0f;
    t[4] = (lg[4] + gb.x) * 2.0f;  t[5] = (lg[5] + gb.y) * 2.0f;
    t[6] = (lg[6] + gb.z) * 2.0f;  t[7] = (lg[7] + gb.w) * 2.0f;

    float mx = t[0];
#pragma unroll
    for (int j = 1; j < F_OUT; ++j) mx = fmaxf(mx, t[j]);
    float s = 0.0f;
#pragma unroll
    for (int j = 0; j < F_OUT; ++j) { t[j] = __expf(t[j] - mx); s += t[j]; }
    float inv = 1.0f / s;

    float4* go = (float4*)(g_out + r * F_OUT);
    go[0] = make_float4(t[0] * inv, t[1] * inv, t[2] * inv, t[3] * inv);
    go[1] = make_float4(t[4] * inv, t[5] * inv, t[6] * inv, t[7] * inv);

    float4* po = (float4*)(p_out + r * F_OUT);
    po[0] = make_float4(pr[0], pr[1], pr[2], pr[3]);
    po[1] = make_float4(pr[4], pr[5], pr[6], pr[7]);
}

extern "C" void kernel_launch(void* const* d_in, const int* in_sizes, int n_in,
                              void* d_out, int out_size, void* d_ws, size_t ws_size,
                              hipStream_t stream) {
    // setup_inputs order:
    // 0 graph, 1 rel_rec, 2 rel_send, 3 gumbel_noise,
    // 4..9  m1_W1,m1_b1,m1_W2,m1_b2,m1_g,m1_beta,
    // 10..15 m2_W1,m2_b1,m2_W2,m2_b2,m2_g,m2_beta,
    // 16 Wi, 17 bi, 18 Wn, 19 bn_b, 20 f1_W, 21 f1_b, 22 f2_W, 23 f2_b, 24 f3_W, 25 f3_b
    const float* gumbel  = (const float*)d_in[3];
    const float* m2_beta = (const float*)d_in[15];
    const float* Wi  = (const float*)d_in[16];
    const float* bi  = (const float*)d_in[17];
    const float* Wn  = (const float*)d_in[18];
    const float* bnb = (const float*)d_in[19];
    const float* f1W = (const float*)d_in[20];
    const float* f1b = (const float*)d_in[21];
    const float* f2W = (const float*)d_in[22];
    const float* f2b = (const float*)d_in[23];
    const float* f3W = (const float*)d_in[24];
    const float* f3b = (const float*)d_in[25];

    float* head  = (float*)d_ws;                       // 16 floats
    float* g_out = (float*)d_out;                      // [B,E,8] gumbel-softmax
    float* p_out = g_out + (long)ROWS * F_OUT;         // [B,E,8] prob

    nri_head_kernel<<<1, 512, 0, stream>>>(m2_beta, Wi, bi, Wn, bnb,
                                           f1W, f1b, f2W, f2b, f3W, f3b, head);

    nri_gumbel_kernel<<<(ROWS + 255) / 256, 256, 0, stream>>>(gumbel, head, g_out, p_out);
}
